// EdgeGraphsNBFNet_81037442941438
// MI455X (gfx1250) — matrix operand order repaired and checked by reading, not compiled, not run on hardware
//
#include <hip/hip_runtime.h>
#include <math.h>

typedef __attribute__((ext_vector_type(16))) __bf16          v16bf;
typedef __attribute__((ext_vector_type(16))) unsigned short  v16u;
typedef __attribute__((ext_vector_type(8)))  float           v8f;

static constexpr int NN = 500000;   // nodes
static constexpr int NE = 8000000;  // edges
static constexpr int NP = 100000;   // pp-edges
static constexpr int D  = 256;      // feature dim
static constexpr int WPAD = 272;    // 256 + 16 elem pad (32B-aligned rows)

__device__ __forceinline__ unsigned short f2bf(float f) {
    unsigned int u = __float_as_uint(f);
    unsigned int r = (u + 0x7FFFu + ((u >> 16) & 1u)) >> 16;   // RNE truncate
    return (unsigned short)r;
}

// ---------------- K0a: zero aggr + stats region of workspace ----------------
__global__ void k_zero_ws(float* ws, int count) {
    int i = blockIdx.x * blockDim.x + threadIdx.x;
    if (i < count) ws[i] = 0.f;
}

// ---------------- K0b: fill output (up_emb rows + zero repr rows) -----------
__global__ void k_init_out(float* __restrict__ out, const float* __restrict__ up,
                           const int* __restrict__ te) {
    long i = (long)blockIdx.x * blockDim.x + threadIdx.x;
    if (i >= (long)150000 * D) return;
    int numUp = te[0] - NP;
    long row = i >> 8;
    out[i] = (row < numUp) ? up[i & 255] : 0.f;
}

// ---------------- K1: edge message + scatter-add --------------------------
__global__ void k_edges(const float* __restrict__ x, const int* __restrict__ ei,
                        const float* __restrict__ ea, const float* __restrict__ We,
                        const float* __restrict__ be, float* __restrict__ aggr) {
    int e = blockIdx.x * blockDim.x + threadIdx.x;
    if (e >= NE) return;
    float w0 = We[0], w1e = We[1], b = be[0];
    int s = ei[e];
    int d = ei[NE + e];
    float m = fmaxf(fmaf(ea[2 * e], w0, fmaf(ea[2 * e + 1], w1e, x[s] + b)), 0.f);
    atomicAdd(&aggr[d], m);
}

// ---------------- K2: z = x + aggr; column sums of relu(z*w1+b1) ----------
__global__ __launch_bounds__(256) void k_z_s1(const float* __restrict__ x,
                                              const float* __restrict__ aggr,
                                              const float* __restrict__ w1,
                                              const float* __restrict__ b1,
                                              float* __restrict__ zbuf,
                                              float* __restrict__ s1) {
    __shared__ float sz[512];
    int tid = threadIdx.x;
    int base = blockIdx.x * 512;
    for (int i = tid; i < 512; i += 256) {
        int n = base + i;
        float zz = 0.f;
        if (n < NN) { zz = x[n] + aggr[n]; zbuf[n] = zz; }
        sz[i] = zz;
    }
    __syncthreads();
    int cnt = NN - base; if (cnt > 512) cnt = 512;
    float w = w1[tid], bb = b1[tid], s = 0.f;
    for (int i = 0; i < cnt; ++i) s += fmaxf(fmaf(sz[i], w, bb), 0.f);
    atomicAdd(&s1[tid], s);
}

// ---------------- K3: mean = (s1 @ w2)/N + b2 -----------------------------
__global__ void k_mean(const float* __restrict__ s1, const float* __restrict__ w2,
                       const float* __restrict__ b2, float* __restrict__ meang) {
    int d = threadIdx.x;
    float acc = 0.f;
    for (int k = 0; k < D; ++k) acc = fmaf(s1[k], w2[k * D + d], acc);
    meang[d] = acc * (1.0f / (float)NN) + b2[d];
}

// ---------------- K5: BN scale/shift --------------------------------------
__global__ void k_bnparam(const float* __restrict__ meang, const float* __restrict__ sumsqg,
                          const float* __restrict__ gamma, const float* __restrict__ beta,
                          float* __restrict__ scaleg, float* __restrict__ shiftg) {
    int d = threadIdx.x;
    float m = meang[d];
    float var = sumsqg[d] * (1.0f / (float)NN) - m * m;
    float sc = gamma[d] * rsqrtf(var + 1e-5f);
    scaleg[d] = sc;
    shiftg[d] = beta[d] - m * sc;
}

// ---------------- K4/K6: bf16 WMMA GEMM, two epilogues --------------------
// mode 0: accumulate sum(h^2) per column.  mode 1: BN+relu+atomic segment-sum.
__global__ __launch_bounds__(256) void k_gemm(const float* __restrict__ zbuf,
                                              const float* __restrict__ w1,
                                              const float* __restrict__ b1,
                                              const float* __restrict__ w2,
                                              const float* __restrict__ b2,
                                              const float* __restrict__ scaleg,
                                              const float* __restrict__ shiftg,
                                              float* __restrict__ sumsq_g,
                                              const int* __restrict__ seg,
                                              const int* __restrict__ te,
                                              float* __restrict__ out, int mode) {
    __shared__ unsigned short sW[64 * WPAD];   // w2^T slice, bf16, padded rows
    __shared__ float sZ[128];
    __shared__ float sw1[D], sb1[D];
    __shared__ float sb2[64], sSc[64], sSh[64], sSq[64];
    __shared__ int   sRow[128];

    const int tid = threadIdx.x;
    const int nb0 = blockIdx.x * 128;          // node base of this block
    const int c0  = blockIdx.y * 64;           // column base of this slice

    // stage w2^T (bf16) into LDS: sW[c][k] = w2[k][c0+c]
    for (int i = tid; i < 64 * 256; i += 256) {
        int c = i & 63, k = i >> 6;
        sW[c * WPAD + k] = f2bf(w2[k * D + c0 + c]);
    }
    sw1[tid] = w1[tid];
    sb1[tid] = b1[tid];
    if (tid < 128) { int n = nb0 + tid; sZ[tid] = (n < NN) ? zbuf[n] : 0.f; }
    if (tid < 64) {
        sb2[tid] = b2[c0 + tid];
        if (mode == 0) { sSq[tid] = 0.f; }
        else           { sSc[tid] = scaleg[c0 + tid]; sSh[tid] = shiftg[c0 + tid]; }
    }
    if (mode == 1 && tid < 128) {
        int n = nb0 + tid;
        sRow[tid] = (n < NN) ? (te[0] - NP + seg[n]) : 0;
    }
    __syncthreads();

    const int lane = tid & 31, w = tid >> 5;
    const int lnN = lane & 15, half = lane >> 4;
    const float z = sZ[w * 16 + lnN];          // A-row scalar for this lane's M

    v8f acc[4];
    #pragma unroll
    for (int t = 0; t < 4; ++t)
        #pragma unroll
        for (int v = 0; v < 8; ++v) acc[t][v] = 0.f;

    // Software-pipelined, fully-unrolled main loop: B fragments for step ks+1
    // are issued before the 4 WMMAs of step ks; full unroll turns the buffer
    // rotation into register renaming (no v_mov copies across a backedge).
    const unsigned short* sWb = &sW[half * 16];   // lane-half K offset baked in
    v16bf bcur[4];
    #pragma unroll
    for (int t = 0; t < 4; ++t)
        bcur[t] = *(const v16bf*)&sWb[(t * 16 + lnN) * WPAD];

    #pragma unroll
    for (int ks = 0; ks < 8; ++ks) {
        const int k0 = ks * 32;
        // Build A fragment (16x32 bf16) in registers from scalar z.
        // Layout: VGPR v -> K = half*16 + 2v
        v16u au;
        #pragma unroll
        for (int v = 0; v < 8; ++v) {
            int k = k0 + half * 16 + v * 2;
            float h0 = fmaxf(fmaf(z, sw1[k],     sb1[k]),     0.f);
            float h1 = fmaxf(fmaf(z, sw1[k + 1], sb1[k + 1]), 0.f);
            au[2 * v]     = f2bf(h0);
            au[2 * v + 1] = f2bf(h1);
        }
        union { v16u u; v16bf b; } cvt; cvt.u = au;
        v16bf a = cvt.b;

        v16bf bnext[4];
        if (ks < 7) {
            #pragma unroll
            for (int t = 0; t < 4; ++t)
                bnext[t] = *(const v16bf*)&sWb[(t * 16 + lnN) * WPAD + k0 + 32];
        }

        #pragma unroll
        for (int t = 0; t < 4; ++t)
            acc[t] = __builtin_amdgcn_wmma_f32_16x16x32_bf16(
                false, a, false, bcur[t], (short)0, acc[t], false, false);

        if (ks < 7) {
            #pragma unroll
            for (int t = 0; t < 4; ++t) bcur[t] = bnext[t];
        }
    }

    // Epilogue. C/D layout: lane col = tile*16 + lnN; VGPR v -> row M = v + half*8.
    const bool full = (nb0 + 128 <= NN);       // uniform: all rows valid
    if (mode == 0) {
        #pragma unroll
        for (int t = 0; t < 4; ++t) {
            int colL = t * 16 + lnN;
            float bb = sb2[colL];
            float s = 0.f;
            if (full) {
                #pragma unroll
                for (int v = 0; v < 8; ++v) {
                    float hv = acc[t][v] + bb;
                    s = fmaf(hv, hv, s);
                }
            } else {
                #pragma unroll
                for (int v = 0; v < 8; ++v) {
                    int node = nb0 + w * 16 + v + half * 8;
                    if (node < NN) { float hv = acc[t][v] + bb; s = fmaf(hv, hv, s); }
                }
            }
            atomicAdd(&sSq[colL], s);
        }
        __syncthreads();
        if (tid < 64) atomicAdd(&sumsq_g[c0 + tid], sSq[tid]);
    } else {
        #pragma unroll
        for (int t = 0; t < 4; ++t) {
            int colL = t * 16 + lnN;
            float bb = sb2[colL], sc = sSc[colL], sh = sSh[colL];
            if (full) {
                #pragma unroll
                for (int v = 0; v < 8; ++v) {
                    int mi = w * 16 + v + half * 8;
                    float y = fmaxf(fmaf(acc[t][v] + bb, sc, sh), 0.f);
                    atomicAdd(&out[(size_t)sRow[mi] * D + c0 + colL], y);
                }
            } else {
                #pragma unroll
                for (int v = 0; v < 8; ++v) {
                    int mi = w * 16 + v + half * 8;
                    int node = nb0 + mi;
                    if (node < NN) {
                        float y = fmaxf(fmaf(acc[t][v] + bb, sc, sh), 0.f);
                        atomicAdd(&out[(size_t)sRow[mi] * D + c0 + colL], y);
                    }
                }
            }
        }
    }
}

extern "C" void kernel_launch(void* const* d_in, const int* in_sizes, int n_in,
                              void* d_out, int out_size, void* d_ws, size_t ws_size,
                              hipStream_t stream) {
    const float* x     = (const float*)d_in[0];
    const int*   ei    = (const int*)  d_in[1];
    const float* ea    = (const float*)d_in[2];
    const int*   seg   = (const int*)  d_in[3];
    const int*   te    = (const int*)  d_in[4];
    const float* We    = (const float*)d_in[5];
    const float* be    = (const float*)d_in[6];
    const float* w1    = (const float*)d_in[7];
    const float* b1    = (const float*)d_in[8];
    const float* w2    = (const float*)d_in[9];
    const float* b2    = (const float*)d_in[10];
    const float* gamma = (const float*)d_in[11];
    const float* beta  = (const float*)d_in[12];
    const float* up    = (const float*)d_in[13];
    float* out = (float*)d_out;

    float* ws     = (float*)d_ws;
    float* aggr   = ws;                 // [NN]
    float* zbuf   = ws + NN;            // [NN]
    float* s1     = ws + 2 * NN;        // [256]
    float* meang  = s1 + 256;           // [256]
    float* sumsqg = meang + 256;        // [256]
    float* scaleg = sumsqg + 256;       // [256]
    float* shiftg = scaleg + 256;       // [256]

    // K0: init workspace (aggr, z, s1..sumsq region) and output
    int zcount = 2 * NN + 1024;
    k_zero_ws<<<(zcount + 255) / 256, 256, 0, stream>>>(ws, zcount);
    long outElems = 150000L * D;
    k_init_out<<<(int)((outElems + 255) / 256), 256, 0, stream>>>(out, up, te);

    // K1: edge scatter
    k_edges<<<(NE + 255) / 256, 256, 0, stream>>>(x, ei, ea, We, be, aggr);

    // K2: z + column sums of h1
    k_z_s1<<<(NN + 511) / 512, 256, 0, stream>>>(x, aggr, w1, b1, zbuf, s1);

    // K3: mean via tiny matvec
    k_mean<<<1, 256, 0, stream>>>(s1, w2, b2, meang);

    // K4: WMMA pass A — sum of squares
    dim3 ggrid((NN + 127) / 128, 4, 1);
    k_gemm<<<ggrid, 256, 0, stream>>>(zbuf, w1, b1, w2, b2, scaleg, shiftg,
                                      sumsqg, seg, te, out, 0);

    // K5: BN scale/shift
    k_bnparam<<<1, 256, 0, stream>>>(meang, sumsqg, gamma, beta, scaleg, shiftg);

    // K6: WMMA pass B — normalize + relu + atomic segment-sum into out
    k_gemm<<<ggrid, 256, 0, stream>>>(zbuf, w1, b1, w2, b2, scaleg, shiftg,
                                      sumsqg, seg, te, out, 1);
}
